// CrossAttention_4226247819633
// MI455X (gfx1250) — matrix operand order repaired and checked by reading, not compiled, not run on hardware
//
#include <hip/hip_runtime.h>
#include <hip/hip_bf16.h>

typedef __attribute__((ext_vector_type(16))) __bf16 v16bf;
typedef __attribute__((ext_vector_type(8)))  __bf16 v8bf;
typedef __attribute__((ext_vector_type(8)))  float  v8f;

#define CC      256
#define HIDD    128
#define NPIX    4096      // 64*64
#define IMGH    256
#define IMGHW   65536     // 256*256
#define KDOWN   2304      // 256*9
#define SCALE_F 0.08838834764831845f

// ---- WMMA fragment helpers (wave32, contiguous k order used here for A and B, 16-bit A 16x32 / B 32x16) ----
// A: lane m=lane&15, kb=(lane>>4)*16 -> elements K = kb..kb+15 (contiguous)
// B: lane n=lane&15, kb=(lane>>4)*16 -> elements K = kb..kb+15 (contiguous)
__device__ __forceinline__ v16bf ld_frag(const __bf16* p0, const __bf16* p1) {
  v8bf lo = *(const v8bf*)p0;
  v8bf hi = *(const v8bf*)p1;
  return __builtin_shufflevector(lo, hi, 0,1,2,3,4,5,6,7,8,9,10,11,12,13,14,15);
}
__device__ __forceinline__ v8f wmma_bf(v16bf a, v16bf b, v8f c) {
  return __builtin_amdgcn_wmma_f32_16x16x32_bf16(false, a, false, b, (short)0, c, false, false);
}

// ---------------- weight packing ----------------
// w_down (O,I,3,3) f32 -> bf16 [O][tap*256+ci] (im2col-K row-major)
__global__ void k_pack_wdown(const float* __restrict__ w, __bf16* __restrict__ dst) {
  int idx = blockIdx.x * 256 + threadIdx.x;          // < 589824
  int o = idx / KDOWN;
  int r = idx - o * KDOWN;
  int tap = r >> 8, ci = r & 255;
  dst[idx] = (__bf16)w[(o * CC + ci) * 9 + tap];
}

__global__ void k_cvt(const float* __restrict__ s, __bf16* __restrict__ d, int n) {
  int i = blockIdx.x * 256 + threadIdx.x;
  if (i < n) d[i] = (__bf16)s[i];
}

// Wfuse_x1 block: w_fuse[f][512+c]
__global__ void k_pack_wfc(const float* __restrict__ wf, __bf16* __restrict__ d) {
  int i = blockIdx.x * 256 + threadIdx.x;            // < 65536
  int f = i >> 8, c = i & 255;
  d[i] = (__bf16)wf[f * 768 + 512 + c];
}

// Ma[t][f][i] = sum_o wfuse[f][br*256+o] * w_up[i][o][t]   (w_up is (I,O,3,3))
__global__ void k_mab(const float* __restrict__ wf, const float* __restrict__ wup,
                      __bf16* __restrict__ Ma, __bf16* __restrict__ Mb) {
  int idx = blockIdx.x * 256 + threadIdx.x;          // < 589824
  int br = blockIdx.y;
  int t = idx >> 16;
  int fi = idx & 65535;
  int f = fi >> 8, i = fi & 255;
  const float* wrow = wf + f * 768 + br * 256;
  const float* urow = wup + (size_t)i * CC * 9 + t;
  float s = 0.f;
  for (int o = 0; o < 256; ++o) s += wrow[o] * urow[o * 9];
  (br ? Mb : Ma)[idx] = (__bf16)s;
}

// b_const[f] = b_fuse[f] + sum_c (wf[f][c] + wf[f][256+c]) * b_up[c]
__global__ void k_bconst(const float* __restrict__ wf, const float* __restrict__ bup,
                         const float* __restrict__ bfuse, float* __restrict__ bc) {
  int f = threadIdx.x;
  float s = bfuse[f];
  const float* wrow = wf + f * 768;
  for (int c = 0; c < 256; ++c) s += (wrow[c] + wrow[256 + c]) * bup[c];
  bc[f] = s;
}

// x1 (C,65536) f32 -> x1T (65536,C) bf16, LDS-tiled transpose
__global__ void k_x1T(const float* __restrict__ x, __bf16* __restrict__ xT) {
  __shared__ float tile[32][33];
  int tx = threadIdx.x & 31, ty = threadIdx.x >> 5;  // ty 0..7
  int p0 = blockIdx.x * 32, c0 = blockIdx.y * 32;
  for (int k = 0; k < 4; ++k)
    tile[ty + k * 8][tx] = x[(size_t)(c0 + ty + k * 8) * IMGHW + p0 + tx];
  __syncthreads();
  for (int k = 0; k < 4; ++k)
    xT[(size_t)(p0 + ty + k * 8) * CC + c0 + tx] = (__bf16)tile[tx][ty + k * 8];
}

// ---------------- down conv: 3x3 stride4 pad1, im2col WMMA ----------------
// out xdT[img][pix][c] bf16.  8 waves = 8 out-channel tiles sharing a double-buffered
// LDS B-slab; next K-step staged (branch-free masked loads) before current WMMA.
__global__ void k_down(const float* __restrict__ x1, const float* __restrict__ x2,
                       const float* __restrict__ x3, const __bf16* __restrict__ wA,
                       const float* __restrict__ bd, __bf16* __restrict__ xdT) {
  const float* x = (blockIdx.z == 0) ? x1 : ((blockIdx.z == 1) ? x2 : x3);
  __bf16* out = xdT + (size_t)blockIdx.z * NPIX * CC;
  __shared__ __align__(16) __bf16 BsT[2][512];       // [buf][n*32+k]
  int tid = threadIdx.x, lane = tid & 31, w = tid >> 5;
  int n15 = lane & 15, hi = lane >> 4;
  int oy = blockIdx.x >> 2, ox0 = (blockIdx.x & 3) * 16;
  int mt = blockIdx.y * 8 + w;
  const __bf16* arow = wA + (size_t)(mt * 16 + n15) * KDOWN + hi * 16;
  int ln = tid & 15, lk = tid >> 4;

  auto stage = [&](int step, int buf) {
    int tap = step >> 3, ci0 = (step & 7) << 5;
    int ky = tap / 3, kx = tap - ky * 3;
    int iy = oy * 4 - 1 + ky;
    int ix = (ox0 + ln) * 4 - 1 + kx;
    float msk = ((iy >= 0) & (ix >= 0)) ? 1.f : 0.f; // branch-free edge handling
    int iyc = iy < 0 ? 0 : iy;
    int ixc = ix < 0 ? 0 : ix;
    size_t base = (size_t)iyc * IMGH + ixc;
    float v0 = x[(size_t)(ci0 + lk) * IMGHW + base] * msk;
    float v1 = x[(size_t)(ci0 + lk + 16) * IMGHW + base] * msk;
    BsT[buf][ln * 32 + lk] = (__bf16)v0;
    BsT[buf][ln * 32 + lk + 16] = (__bf16)v1;
  };

  v8f acc; for (int v = 0; v < 8; ++v) acc[v] = 0.f;
  stage(0, 0);
  __syncthreads();
  for (int step = 0; step < 72; ++step) {
    int buf = step & 1;
    if (step + 1 < 72) stage(step + 1, buf ^ 1);     // issue global loads early
    v16bf a = ld_frag(arow + step * 32, arow + step * 32 + 8);
    v16bf b = ld_frag(&BsT[buf][n15 * 32 + hi * 16], &BsT[buf][n15 * 32 + hi * 16 + 8]);
    acc = wmma_bf(a, b, acc);
    __syncthreads();                                 // single barrier per K-step
  }
  v8bf o;
  for (int v = 0; v < 8; ++v) o[v] = (__bf16)(acc[v] + bd[mt * 16 + hi * 8 + v]);
  int pix = oy * 64 + ox0 + n15;
  *(v8bf*)(out + (size_t)pix * CC + mt * 16 + hi * 8) = o;
}

// ---------------- 1x1 convs (q/k/v): W(MxC) @ xdT^T ----------------
__global__ void k_qkv(const __bf16* __restrict__ W, const __bf16* __restrict__ xdT,
                      const float* __restrict__ bias, __bf16* __restrict__ out,
                      int M, int transposed) {
  int tid = threadIdx.x, lane = tid & 31, w = tid >> 5;
  int n15 = lane & 15, hi = lane >> 4;
  int n0 = (blockIdx.x * 8 + w) * 16;
  int m0 = blockIdx.y * 16;
  const __bf16* arow = W + (size_t)(m0 + n15) * CC + hi * 16;
  const __bf16* brow = xdT + (size_t)(n0 + n15) * CC + hi * 16;
  v8f acc; for (int v = 0; v < 8; ++v) acc[v] = 0.f;
  for (int s = 0; s < 8; ++s) {
    acc = wmma_bf(ld_frag(arow + s * 32, arow + s * 32 + 8),
                  ld_frag(brow + s * 32, brow + s * 32 + 8), acc);
  }
  if (transposed) {                                  // out[pix][m] (qT / kT)
    v8bf o;
    for (int v = 0; v < 8; ++v) o[v] = (__bf16)(acc[v] + bias[m0 + hi * 8 + v]);
    *(v8bf*)(out + (size_t)(n0 + n15) * M + m0 + hi * 8) = o;
  } else {                                           // out[m][pix] (v)
    for (int v = 0; v < 8; ++v) {
      int m = m0 + hi * 8 + v;
      out[(size_t)m * NPIX + n0 + n15] = (__bf16)(acc[v] + bias[m]);
    }
  }
}

// ---------------- flash attention: one wave owns 16 queries, full 256-ch acc ----------------
__global__ void k_flash(const __bf16* __restrict__ qT,
                        const __bf16* __restrict__ kaT, const __bf16* __restrict__ kbT,
                        const __bf16* __restrict__ va, const __bf16* __restrict__ vb,
                        __bf16* __restrict__ fTa, __bf16* __restrict__ fTb) {
  int lane = threadIdx.x, n15 = lane & 15, hi = lane >> 4;
  int br = blockIdx.y;
  const __bf16* kT = br ? kbT : kaT;
  const __bf16* V  = br ? vb : va;
  __bf16* fT = br ? fTb : fTa;
  int i0 = blockIdx.x * 16;
  v16bf qf[4];
  const __bf16* qrow = qT + (size_t)(i0 + n15) * HIDD + hi * 16;
  for (int s = 0; s < 4; ++s) qf[s] = ld_frag(qrow + s * 32, qrow + s * 32 + 8);
  v8f acc[16];
  for (int ct = 0; ct < 16; ++ct) for (int v = 0; v < 8; ++v) acc[ct][v] = 0.f;
  float mrow = -1e30f, lrow = 0.f;
  __shared__ float Sbuf[512];
  __shared__ __align__(16) __bf16 Pbuf[512];
  __shared__ float abuf[16];
  __shared__ float lbuf[16];
  for (int jb = 0; jb < 128; ++jb) {
    int j0 = jb * 32;
    if (jb + 2 < 128) {                              // -> global_prefetch_b8
      __builtin_prefetch(kT + (size_t)(j0 + 64 + n15) * HIDD, 0, 1);
      __builtin_prefetch(V + (size_t)n15 * NPIX + j0 + 64, 0, 1);
    }
    // S = Q^T K  (two 16x16 tiles)
    for (int jt = 0; jt < 2; ++jt) {
      v8f s; for (int v = 0; v < 8; ++v) s[v] = 0.f;
      const __bf16* kb = kT + (size_t)(j0 + jt * 16 + n15) * HIDD + hi * 16;
      for (int ss = 0; ss < 4; ++ss)
        s = wmma_bf(qf[ss], ld_frag(kb + ss * 32, kb + ss * 32 + 8), s);
      for (int v = 0; v < 8; ++v) Sbuf[(v + 8 * hi) * 32 + jt * 16 + n15] = s[v];
    }
    __syncthreads();
    if (lane < 16) {                                 // row-owner online softmax
      float mx = mrow;
      for (int j = 0; j < 32; ++j) mx = fmaxf(mx, Sbuf[lane * 32 + j] * SCALE_F);
      float alpha = __expf(mrow - mx);
      float sum = 0.f;
      for (int j = 0; j < 32; ++j) {
        float p = __expf(Sbuf[lane * 32 + j] * SCALE_F - mx);
        sum += p;
        Pbuf[lane * 32 + j] = (__bf16)p;
      }
      lrow = lrow * alpha + sum;
      mrow = mx;
      abuf[lane] = alpha;
    }
    __syncthreads();
    float alpha = abuf[n15];                         // feat column i == lane&15
    for (int ct = 0; ct < 16; ++ct)
      for (int v = 0; v < 8; ++v) acc[ct][v] *= alpha;
    v16bf pf = ld_frag(&Pbuf[n15 * 32 + hi * 16], &Pbuf[n15 * 32 + hi * 16 + 8]);
    for (int ct = 0; ct < 16; ++ct) {                // feat += V * P^T
      const __bf16* vr = V + (size_t)(ct * 16 + n15) * NPIX + j0 + hi * 16;
      acc[ct] = wmma_bf(ld_frag(vr, vr + 8), pf, acc[ct]);
    }
    __syncthreads();
  }
  if (lane < 16) lbuf[lane] = lrow;
  __syncthreads();
  float inv = 1.f / lbuf[n15];
  __bf16* orow = fT + (size_t)(i0 + n15) * CC;
  for (int ct = 0; ct < 16; ++ct) {
    v8bf o;
    for (int v = 0; v < 8; ++v) o[v] = (__bf16)(acc[ct][v] * inv);
    *(v8bf*)(orow + ct * 16 + hi * 8) = o;
  }
}

// ---------------- fused convT + concat + 1x1 fuse ----------------
// out[f,oy,ox] = bconst[f] + Wfc@x1[:,p] + (ry<3&&rx<3 ? Ma[t]@feat_a + Mb[t]@feat_b : 0)
__global__ void k_fuse(const __bf16* __restrict__ wfc, const __bf16* __restrict__ Ma,
                       const __bf16* __restrict__ Mb, const __bf16* __restrict__ x1T,
                       const __bf16* __restrict__ fTa, const __bf16* __restrict__ fTb,
                       const float* __restrict__ bconst, float* __restrict__ out) {
  int tid = threadIdx.x, lane = tid & 31, w = tid >> 5;
  int n15 = lane & 15, hi = lane >> 4;
  int cls = blockIdx.z, ry = cls >> 2, rx = cls & 3;
  int iy = blockIdx.x >> 2, ix0 = (blockIdx.x & 3) * 16;
  int m0 = (blockIdx.y * 8 + w) * 16;
  int oy = iy * 4 + ry;
  int ox = (ix0 + n15) * 4 + rx;
  v8f acc; for (int v = 0; v < 8; ++v) acc[v] = 0.f;
  const __bf16* a1 = wfc + (size_t)(m0 + n15) * CC + hi * 16;
  const __bf16* b1 = x1T + ((size_t)oy * IMGH + ox) * CC + hi * 16;
  for (int s = 0; s < 8; ++s)
    acc = wmma_bf(ld_frag(a1 + s * 32, a1 + s * 32 + 8),
                  ld_frag(b1 + s * 32, b1 + s * 32 + 8), acc);
  if (ry < 3 && rx < 3) {                            // uniform per block
    int tap = ry * 3 + rx;
    size_t fp = (size_t)(iy * 64 + ix0 + n15) * CC;
    const __bf16* a2 = Ma + (size_t)tap * 65536 + (size_t)(m0 + n15) * CC + hi * 16;
    const __bf16* b2 = fTa + fp + hi * 16;
    for (int s = 0; s < 8; ++s)
      acc = wmma_bf(ld_frag(a2 + s * 32, a2 + s * 32 + 8),
                    ld_frag(b2 + s * 32, b2 + s * 32 + 8), acc);
    const __bf16* a3 = Mb + (size_t)tap * 65536 + (size_t)(m0 + n15) * CC + hi * 16;
    const __bf16* b3 = fTb + fp + hi * 16;
    for (int s = 0; s < 8; ++s)
      acc = wmma_bf(ld_frag(a3 + s * 32, a3 + s * 32 + 8),
                    ld_frag(b3 + s * 32, b3 + s * 32 + 8), acc);
  }
  for (int v = 0; v < 8; ++v) {
    int m = m0 + hi * 8 + v;
    out[(size_t)m * IMGHW + (size_t)oy * IMGH + ox] = acc[v] + bconst[m];
  }
}

// ---------------- host ----------------
extern "C" void kernel_launch(void* const* d_in, const int* in_sizes, int n_in,
                              void* d_out, int out_size, void* d_ws, size_t ws_size,
                              hipStream_t stream) {
  const float* x1     = (const float*)d_in[0];
  const float* x2     = (const float*)d_in[1];
  const float* x3     = (const float*)d_in[2];
  const float* w_down = (const float*)d_in[3];
  const float* b_down = (const float*)d_in[4];
  const float* w_q    = (const float*)d_in[5];
  const float* b_q    = (const float*)d_in[6];
  const float* w_k    = (const float*)d_in[7];
  const float* b_k    = (const float*)d_in[8];
  const float* w_v    = (const float*)d_in[9];
  const float* b_v    = (const float*)d_in[10];
  const float* w_up   = (const float*)d_in[11];
  const float* b_up   = (const float*)d_in[12];
  const float* w_fuse = (const float*)d_in[13];
  const float* b_fuse = (const float*)d_in[14];
  float* out = (float*)d_out;

  char* ws = (char*)d_ws;
  size_t off = 0;
  auto alloc = [&](size_t bytes) -> char* {
    size_t o = (off + 255) & ~(size_t)255;
    off = o + bytes;
    return ws + o;
  };
  __bf16* wdown_bf = (__bf16*)alloc((size_t)589824 * 2);
  __bf16* wq_bf    = (__bf16*)alloc((size_t)32768 * 2);
  __bf16* wk_bf    = (__bf16*)alloc((size_t)32768 * 2);
  __bf16* wv_bf    = (__bf16*)alloc((size_t)65536 * 2);
  __bf16* wfc_bf   = (__bf16*)alloc((size_t)65536 * 2);
  __bf16* Ma_bf    = (__bf16*)alloc((size_t)589824 * 2);
  __bf16* Mb_bf    = (__bf16*)alloc((size_t)589824 * 2);
  float*  bconst   = (float*)alloc((size_t)256 * 4);
  __bf16* xdT      = (__bf16*)alloc((size_t)3 * NPIX * CC * 2);
  __bf16* qT       = (__bf16*)alloc((size_t)NPIX * HIDD * 2);
  __bf16* kaT      = (__bf16*)alloc((size_t)NPIX * HIDD * 2);
  __bf16* kbT      = (__bf16*)alloc((size_t)NPIX * HIDD * 2);
  __bf16* va       = (__bf16*)alloc((size_t)CC * NPIX * 2);
  __bf16* vb       = (__bf16*)alloc((size_t)CC * NPIX * 2);
  __bf16* fTa      = (__bf16*)alloc((size_t)NPIX * CC * 2);
  __bf16* fTb      = (__bf16*)alloc((size_t)NPIX * CC * 2);
  __bf16* x1T      = (__bf16*)alloc((size_t)IMGHW * CC * 2);
  if (off > ws_size) return;  // workspace insufficient: do nothing (deterministic)

  // weight packing / precomputation
  k_pack_wdown<<<2304, 256, 0, stream>>>(w_down, wdown_bf);
  k_cvt<<<128, 256, 0, stream>>>(w_q, wq_bf, 32768);
  k_cvt<<<128, 256, 0, stream>>>(w_k, wk_bf, 32768);
  k_cvt<<<256, 256, 0, stream>>>(w_v, wv_bf, 65536);
  k_pack_wfc<<<256, 256, 0, stream>>>(w_fuse, wfc_bf);
  k_mab<<<dim3(2304, 2), 256, 0, stream>>>(w_fuse, w_up, Ma_bf, Mb_bf);
  k_bconst<<<1, 256, 0, stream>>>(w_fuse, b_up, b_fuse, bconst);
  k_x1T<<<dim3(2048, 8), 256, 0, stream>>>(x1, x1T);

  // down-convs (3 images in one grid)
  k_down<<<dim3(256, 2, 3), 256, 0, stream>>>(x1, x2, x3, wdown_bf, b_down, xdT);

  // q/k/v 1x1 convs
  __bf16* x1dT = xdT;
  __bf16* x2dT = xdT + (size_t)NPIX * CC;
  __bf16* x3dT = xdT + (size_t)2 * NPIX * CC;
  k_qkv<<<dim3(32, 8),  256, 0, stream>>>(wq_bf, x1dT, b_q, qT,  HIDD, 1);
  k_qkv<<<dim3(32, 8),  256, 0, stream>>>(wk_bf, x2dT, b_k, kaT, HIDD, 1);
  k_qkv<<<dim3(32, 8),  256, 0, stream>>>(wk_bf, x3dT, b_k, kbT, HIDD, 1);
  k_qkv<<<dim3(32, 16), 256, 0, stream>>>(wv_bf, x2dT, b_v, va,  CC,   0);
  k_qkv<<<dim3(32, 16), 256, 0, stream>>>(wv_bf, x3dT, b_v, vb,  CC,   0);

  // two flash attentions (branch = blockIdx.y)
  k_flash<<<dim3(256, 2), 32, 0, stream>>>(qT, kaT, kbT, va, vb, fTa, fTb);

  // fused convT + concat + 1x1 fuse -> output
  k_fuse<<<dim3(256, 2, 16), 256, 0, stream>>>(wfc_bf, Ma_bf, Mb_bf, x1T, fTa, fTb,
                                               bconst, out);
}
